// CausalSelfAttention_481036337201
// MI455X (gfx1250) — compile-verified, matrix operand order checked by reading, not run on hardware
//
#include <hip/hip_runtime.h>
#include <cstdint>
#include <cstddef>

// Problem shape (fixed by reference)
#define B_  4
#define S_  2048
#define D_  1024
#define H_  16
#define DH_ 64

typedef __attribute__((ext_vector_type(16))) __bf16 bf16x16;
typedef __attribute__((ext_vector_type(8)))  float  f32x8;

// TDM availability / arity guard (ROCm7.2 clang-22: 5 args; therock clang-23: 6 args)
#if defined(__has_builtin)
# if __has_builtin(__builtin_amdgcn_tensor_load_to_lds) && \
     __has_builtin(__builtin_amdgcn_s_wait_tensorcnt)
#  define HAVE_TDM 1
# endif
#endif
#ifndef HAVE_TDM
# define HAVE_TDM 0
#endif

// ---------------- workspace layout (bytes) ----------------
#define XB_OFF   ((size_t)0)
#define WQT_OFF  (XB_OFF  + (size_t)B_*S_*D_*2)
#define WKT_OFF  (WQT_OFF + (size_t)D_*D_*2)
#define WVT_OFF  (WKT_OFF + (size_t)D_*D_*2)
#define WOT_OFF  (WVT_OFF + (size_t)D_*D_*2)
#define Q_OFF    (WOT_OFF + (size_t)D_*DH_*2)
#define K_OFF    (Q_OFF   + (size_t)B_*H_*S_*DH_*2)
#define VT_OFF   (K_OFF   + (size_t)B_*H_*S_*DH_*2)
#define ZS_OFF   (VT_OFF  + (size_t)B_*H_*S_*DH_*2)

// ---------------- helpers ----------------
__device__ __forceinline__ __bf16 f2bf(float x) {
  union { float f; uint32_t u; } a; a.f = x;
  uint32_t r = a.u + 0x7FFFu + ((a.u >> 16) & 1u);   // round-to-nearest-even
  union { uint16_t s; __bf16 b; } o; o.s = (uint16_t)(r >> 16);
  return o.b;
}

// ds_swizzle xor reductions within groups of 16 lanes (wave32)
#define DSWZ(v, off) __int_as_float(__builtin_amdgcn_ds_swizzle(__float_as_int(v), (off)))
__device__ __forceinline__ float redmax16(float v) {
  v = fmaxf(v, DSWZ(v, 0x041F));
  v = fmaxf(v, DSWZ(v, 0x081F));
  v = fmaxf(v, DSWZ(v, 0x101F));
  v = fmaxf(v, DSWZ(v, 0x201F));
  return v;
}
__device__ __forceinline__ float redsum16(float v) {
  v += DSWZ(v, 0x041F);
  v += DSWZ(v, 0x081F);
  v += DSWZ(v, 0x101F);
  v += DSWZ(v, 0x201F);
  return v;
}

// 16x32 bf16 A-fragment gather (also for pre-transposed B operands).
__device__ __forceinline__ bf16x16 load_frag(const uint32_t* rowptr32, int hi) {
  union { bf16x16 v; uint32_t u[8]; } f;
#pragma unroll
  for (int j = 0; j < 4; ++j) {
    f.u[j]     = rowptr32[4 * hi + j];
    f.u[4 + j] = rowptr32[8 + 4 * hi + j];
  }
  return f.v;
}

__device__ __forceinline__ bf16x16 load_frag_f32(const float* row, int hi, int kbase) {
  union { bf16x16 v; __bf16 h[16]; } f;
#pragma unroll
  for (int j = 0; j < 4; ++j) {
    int k  = kbase + 8 * hi + 2 * j;
    int k2 = kbase + 16 + 8 * hi + 2 * j;
    f.h[2 * j]     = f2bf(row[k]);
    f.h[2 * j + 1] = f2bf(row[k + 1]);
    f.h[8 + 2 * j]     = f2bf(row[k2]);
    f.h[8 + 2 * j + 1] = f2bf(row[k2 + 1]);
  }
  return f.v;
}

#define WMMA_BF16(a, b, c) \
  __builtin_amdgcn_wmma_f32_16x16x32_bf16(false, (a), false, (b), (short)0, (c), false, false)

#if HAVE_TDM
typedef __attribute__((ext_vector_type(4))) unsigned int u32x4;
typedef __attribute__((ext_vector_type(8))) int i32x8;
typedef __attribute__((ext_vector_type(4))) int i32x4;

// Issue a TDM copy of a 2-D bf16 tile (tile_d0 contiguous elems x tile_d1 rows,
// row stride stride0 elems) from global memory into LDS (packed rows).
// D# layout per CDNA5 ISA 8.3/8.4: group0 = {flags, lds_addr, global_addr, type=2},
// group1 = {data_size=2B, tensor dims (huge -> no OOB), tile dims, strides}.
// All inputs must be wave-uniform (descriptor lives in SGPRs).
__device__ __forceinline__ void tdm_load_2d(const void* gaddr, uint32_t lds_off,
                                            uint32_t tile_d0, uint32_t tile_d1,
                                            uint64_t stride0) {
  const uint64_t ga = (uint64_t)(uintptr_t)gaddr;
  const uint32_t td0 = 0x40000000u;        // tensor_dim0 (huge)
  const uint32_t td1 = 0x00100000u;        // tensor_dim1 (huge)
  u32x4 g0;
  g0[0] = 1u;                                              // count=1, user mode
  g0[1] = lds_off;                                         // lds_addr (bytes)
  g0[2] = (uint32_t)ga;                                    // global_addr[31:0]
  g0[3] = (uint32_t)((ga >> 32) & 0x01FFFFFFu) | 0x80000000u; // addr[56:32] | type=2
  i32x8 g1;
  g1[0] = (int)(1u << 16);                                 // data_size=1 -> 2 bytes
  g1[1] = (int)((td0 & 0xFFFFu) << 16);                    // tensor_dim0[15:0]
  g1[2] = (int)((td0 >> 16) | ((td1 & 0xFFFFu) << 16));    // td0[31:16] | td1[15:0]
  g1[3] = (int)((td1 >> 16) | (tile_d0 << 16));            // td1[31:16] | tile_dim0
  g1[4] = (int)(tile_d1 & 0xFFFFu);                        // tile_dim1, tile_dim2=0
  g1[5] = (int)(uint32_t)(stride0 & 0xFFFFFFFFu);          // dim0_stride[31:0]
  g1[6] = (int)(uint32_t)((stride0 >> 32) & 0xFFFFu);      // dim0_stride[47:32]
  g1[7] = 0;
  const i32x4 gz = {0, 0, 0, 0};
#if __clang_major__ >= 23
  const i32x8 gz8 = {0, 0, 0, 0, 0, 0, 0, 0};
  __builtin_amdgcn_tensor_load_to_lds(g0, g1, gz, gz, gz8, 0);
#else
  __builtin_amdgcn_tensor_load_to_lds(g0, g1, gz, gz, 0);
#endif
}
#endif // HAVE_TDM

// ---------------- prep kernels ----------------
__global__ void cast_f32_bf16_kernel(const float* __restrict__ in, __bf16* __restrict__ out, int n) {
  int i = blockIdx.x * blockDim.x + threadIdx.x;
  if (i < n) out[i] = f2bf(in[i]);
}

__global__ void transpose_cast_kernel(const float* __restrict__ in, __bf16* __restrict__ out,
                                      int rows, int cols) {
  int i = blockIdx.x * blockDim.x + threadIdx.x;
  if (i < rows * cols) {
    int r = i / cols, c = i % cols;
    out[(size_t)c * rows + r] = f2bf(in[i]);
  }
}

__global__ void zero_f32_kernel(float* __restrict__ p, int n) {
  int i = blockIdx.x * blockDim.x + threadIdx.x;
  if (i < n) p[i] = 0.0f;
}

// ---------------- QKV projection: [BS,D] @ [D,D] + bias, 2x4 register blocking ----
__global__ __launch_bounds__(32)
void qkv_proj_kernel(const __bf16* __restrict__ xb,
                     const __bf16* __restrict__ WQt, const __bf16* __restrict__ WKt,
                     const __bf16* __restrict__ WVt,
                     const float* __restrict__ bQ, const float* __restrict__ bK,
                     const float* __restrict__ bV,
                     __bf16* __restrict__ Qb, __bf16* __restrict__ Kb, __bf16* __restrict__ Vt) {
  const int lane = threadIdx.x & 31;
  const int hi = lane >> 4;
  const int mn = lane & 15;
  const int row0 = blockIdx.x * 32;
  const int n0   = blockIdx.y * 64;
  const int which = blockIdx.z;

  const __bf16* Wt  = (which == 0) ? WQt : ((which == 1) ? WKt : WVt);
  const float* bias = (which == 0) ? bQ  : ((which == 1) ? bK  : bV);

  const uint32_t* arow0 = (const uint32_t*)(xb + (size_t)(row0 + mn) * D_);
  const uint32_t* arow1 = (const uint32_t*)(xb + (size_t)(row0 + 16 + mn) * D_);
  const uint32_t* brow0 = (const uint32_t*)(Wt + (size_t)(n0 + 0 * 16 + mn) * D_);
  const uint32_t* brow1 = (const uint32_t*)(Wt + (size_t)(n0 + 1 * 16 + mn) * D_);
  const uint32_t* brow2 = (const uint32_t*)(Wt + (size_t)(n0 + 2 * 16 + mn) * D_);
  const uint32_t* brow3 = (const uint32_t*)(Wt + (size_t)(n0 + 3 * 16 + mn) * D_);

  f32x8 acc[2][4] = {};
  for (int k0 = 0; k0 < D_ / 2; k0 += 16) {
    bf16x16 a0 = load_frag(arow0 + k0, hi);
    bf16x16 a1 = load_frag(arow1 + k0, hi);
    bf16x16 b0 = load_frag(brow0 + k0, hi);
    bf16x16 b1 = load_frag(brow1 + k0, hi);
    bf16x16 b2 = load_frag(brow2 + k0, hi);
    bf16x16 b3 = load_frag(brow3 + k0, hi);
    acc[0][0] = WMMA_BF16(a0, b0, acc[0][0]);
    acc[0][1] = WMMA_BF16(a0, b1, acc[0][1]);
    acc[0][2] = WMMA_BF16(a0, b2, acc[0][2]);
    acc[0][3] = WMMA_BF16(a0, b3, acc[0][3]);
    acc[1][0] = WMMA_BF16(a1, b0, acc[1][0]);
    acc[1][1] = WMMA_BF16(a1, b1, acc[1][1]);
    acc[1][2] = WMMA_BF16(a1, b2, acc[1][2]);
    acc[1][3] = WMMA_BF16(a1, b3, acc[1][3]);
  }

#pragma unroll
  for (int nt = 0; nt < 4; ++nt) {
    const int nglob = n0 + nt * 16 + mn;
    const float bn = bias[nglob];
    const int h  = nglob >> 6;
    const int dh = nglob & (DH_ - 1);
#pragma unroll
    for (int mt = 0; mt < 2; ++mt) {
#pragma unroll
      for (int r = 0; r < 8; ++r) {
        const int row = row0 + mt * 16 + r + 8 * hi;
        const int bb = row / S_, s = row % S_;
        const __bf16 v = f2bf(acc[mt][nt][r] + bn);
        if (which == 0)
          Qb[(((size_t)bb * H_ + h) * S_ + s) * DH_ + dh] = v;
        else if (which == 1)
          Kb[(((size_t)bb * H_ + h) * S_ + s) * DH_ + dh] = v;
        else
          Vt[(((size_t)bb * H_ + h) * DH_ + dh) * S_ + s] = v;   // V transposed
      }
    }
  }
}

// ---------------- flash attention per (b,h, 32-query block) ----------------
// 1 wave/block; K/V tiles staged into LDS by the Tensor Data Mover,
// double-buffered: issue step N+1, s_wait_tensorcnt, consume step N.
__global__ __launch_bounds__(32)
void attn_kernel(const __bf16* __restrict__ Qb, const __bf16* __restrict__ Kb,
                 const __bf16* __restrict__ Vt, float* __restrict__ Zsum) {
#if HAVE_TDM
  __shared__ __align__(16) __bf16 KbufL[2][32 * 64];   // 32 keys x 64 dh, row-major
  __shared__ __align__(16) __bf16 VbufL[2][64 * 32];   // 64 dh rows x 32 keys
#endif
  __shared__ __align__(16) __bf16 Pbuf[2][16 * 32];

  const int lane = threadIdx.x & 31;
  const int hi = lane >> 4;
  const int mn = lane & 15;
  const int qbase = blockIdx.x * 32;
  const int bh = blockIdx.y;

  const __bf16* Qh = Qb + (size_t)bh * S_ * DH_;
  const __bf16* Kh = Kb + (size_t)bh * S_ * DH_;
  const __bf16* Vh = Vt + (size_t)bh * DH_ * S_;

  const uint32_t* qrow0 = (const uint32_t*)(Qh + (size_t)(qbase + mn) * DH_);
  const uint32_t* qrow1 = (const uint32_t*)(Qh + (size_t)(qbase + 16 + mn) * DH_);
  const bf16x16 qa[2][2] = {
    { load_frag(qrow0, hi), load_frag(qrow0 + 16, hi) },
    { load_frag(qrow1, hi), load_frag(qrow1 + 16, hi) },
  };

  f32x8 o[2][4] = {};
  float mrow[2][8], lrow[2][8];
#pragma unroll
  for (int t = 0; t < 2; ++t)
#pragma unroll
    for (int r = 0; r < 8; ++r) { mrow[t][r] = -1e30f; lrow[t][r] = 0.0f; }

  const int nsteps = qbase / 32 + 1;

#if HAVE_TDM
  // prologue: stage step 0 (K tile contiguous 1-D, V tile 2-D strided)
  tdm_load_2d(Kh, (uint32_t)(uintptr_t)&KbufL[0][0], 2048, 0, 2048);
  tdm_load_2d(Vh, (uint32_t)(uintptr_t)&VbufL[0][0], 32, 64, (uint64_t)S_);
#endif

  for (int step = 0; step < nsteps; ++step) {
    const int k0 = step * 32;
#if HAVE_TDM
    const int cur = step & 1;
    if (step + 1 < nsteps) {
      const int k1 = k0 + 32;
      tdm_load_2d(Kh + (size_t)k1 * DH_, (uint32_t)(uintptr_t)&KbufL[cur ^ 1][0], 2048, 0, 2048);
      tdm_load_2d(Vh + k1,               (uint32_t)(uintptr_t)&VbufL[cur ^ 1][0], 32, 64, (uint64_t)S_);
      __builtin_amdgcn_s_wait_tensorcnt(2);   // current buffers complete
    } else {
      __builtin_amdgcn_s_wait_tensorcnt(0);
    }
    __syncthreads();
    const uint32_t* krow0 = (const uint32_t*)&KbufL[cur][(size_t)mn * 64];
    const uint32_t* krow1 = (const uint32_t*)&KbufL[cur][(size_t)(16 + mn) * 64];
#else
    if (step + 1 < nsteps) {
      __builtin_prefetch((const void*)(Kh + (size_t)(k0 + 32 + mn) * DH_), 0, 1);
      __builtin_prefetch((const void*)(Kh + (size_t)(k0 + 48 + mn) * DH_), 0, 1);
      __builtin_prefetch((const void*)(Vh + (size_t)(0 * 16 + mn) * S_ + k0 + 32), 0, 1);
      __builtin_prefetch((const void*)(Vh + (size_t)(2 * 16 + mn) * S_ + k0 + 32), 0, 1);
    }
    const uint32_t* krow0 = (const uint32_t*)(Kh + (size_t)(k0 + mn) * DH_);
    const uint32_t* krow1 = (const uint32_t*)(Kh + (size_t)(k0 + 16 + mn) * DH_);
#endif

    bf16x16 kb00 = load_frag(krow0, hi);
    bf16x16 kb01 = load_frag(krow0 + 16, hi);
    bf16x16 kb10 = load_frag(krow1, hi);
    bf16x16 kb11 = load_frag(krow1 + 16, hi);

    f32x8 s[2][2] = {};
#pragma unroll
    for (int t = 0; t < 2; ++t) {
      s[t][0] = WMMA_BF16(qa[t][0], kb00, s[t][0]);
      s[t][0] = WMMA_BF16(qa[t][1], kb01, s[t][0]);
      s[t][1] = WMMA_BF16(qa[t][0], kb10, s[t][1]);
      s[t][1] = WMMA_BF16(qa[t][1], kb11, s[t][1]);
    }

    // scale 1/sqrt(64), causal mask, online softmax over the 32 key columns
#pragma unroll
    for (int t = 0; t < 2; ++t) {
#pragma unroll
      for (int r = 0; r < 8; ++r) {
        const int q = qbase + 16 * t + r + 8 * hi;
        float v0 = s[t][0][r] * 0.125f;
        float v1 = s[t][1][r] * 0.125f;
        if (k0 + mn > q)      v0 = -1e30f;
        if (k0 + 16 + mn > q) v1 = -1e30f;
        const float rmax = redmax16(fmaxf(v0, v1));
        const float mnew = fmaxf(mrow[t][r], rmax);
        const float p0 = __expf(v0 - mnew);
        const float p1 = __expf(v1 - mnew);
        const float rsum = redsum16(p0 + p1);
        const float alpha = __expf(mrow[t][r] - mnew);
        lrow[t][r] = lrow[t][r] * alpha + rsum;
        mrow[t][r] = mnew;
        o[t][0][r] *= alpha; o[t][1][r] *= alpha;
        o[t][2][r] *= alpha; o[t][3][r] *= alpha;
        const int prow = r + 8 * hi;
        Pbuf[t][prow * 32 + mn]      = f2bf(p0);
        Pbuf[t][prow * 32 + 16 + mn] = f2bf(p1);
      }
    }
    __syncthreads();

    const bf16x16 pa0 = load_frag((const uint32_t*)&Pbuf[0][mn * 32], hi);
    const bf16x16 pa1 = load_frag((const uint32_t*)&Pbuf[1][mn * 32], hi);
#pragma unroll
    for (int nt = 0; nt < 4; ++nt) {
#if HAVE_TDM
      const uint32_t* vr = (const uint32_t*)&VbufL[cur][(size_t)(nt * 16 + mn) * 32];
#else
      const uint32_t* vr = (const uint32_t*)(Vh + (size_t)(nt * 16 + mn) * S_ + k0);
#endif
      const bf16x16 vb = load_frag(vr, hi);
      o[0][nt] = WMMA_BF16(pa0, vb, o[0][nt]);
      o[1][nt] = WMMA_BF16(pa1, vb, o[1][nt]);
    }
    __syncthreads();
  }

  // finalize: divide by row sums; sum over heads into Zsum via f32 atomics
  const int bb = bh / H_;
  float* Zb = Zsum + (size_t)bb * S_ * DH_;
#pragma unroll
  for (int t = 0; t < 2; ++t) {
#pragma unroll
    for (int r = 0; r < 8; ++r) {
      const float invl = 1.0f / lrow[t][r];
      const size_t rowoff = (size_t)(qbase + 16 * t + r + 8 * hi) * DH_;
      atomicAdd(&Zb[rowoff + 0 * 16 + mn], o[t][0][r] * invl);
      atomicAdd(&Zb[rowoff + 1 * 16 + mn], o[t][1][r] * invl);
      atomicAdd(&Zb[rowoff + 2 * 16 + mn], o[t][2][r] * invl);
      atomicAdd(&Zb[rowoff + 3 * 16 + mn], o[t][3][r] * invl);
    }
  }
}

// ---------------- output projection: Zsum[BS,64] @ WO[64,D] + H*bO, 1x4 blocking ----
__global__ __launch_bounds__(32)
void out_proj_kernel(const float* __restrict__ Zsum, const __bf16* __restrict__ WOt,
                     const float* __restrict__ bO, float* __restrict__ out) {
  const int lane = threadIdx.x & 31;
  const int hi = lane >> 4;
  const int mn = lane & 15;
  const int row0 = blockIdx.x * 16;
  const int n0   = blockIdx.y * 64;

  const float* zrow = Zsum + (size_t)(row0 + mn) * DH_;
  bf16x16 a0 = load_frag_f32(zrow, hi, 0);
  bf16x16 a1 = load_frag_f32(zrow, hi, 32);

  f32x8 acc[4] = {};
#pragma unroll
  for (int nt = 0; nt < 4; ++nt) {
    const uint32_t* brow = (const uint32_t*)(WOt + (size_t)(n0 + nt * 16 + mn) * DH_);
    acc[nt] = WMMA_BF16(a0, load_frag(brow, hi), acc[nt]);
    acc[nt] = WMMA_BF16(a1, load_frag(brow + 16, hi), acc[nt]);
  }

#pragma unroll
  for (int nt = 0; nt < 4; ++nt) {
    const float bo = (float)H_ * bO[n0 + nt * 16 + mn];
#pragma unroll
    for (int r = 0; r < 8; ++r)
      out[(size_t)(row0 + r + 8 * hi) * D_ + n0 + nt * 16 + mn] = acc[nt][r] + bo;
  }
}

// ---------------- launch ----------------
extern "C" void kernel_launch(void* const* d_in, const int* in_sizes, int n_in,
                              void* d_out, int out_size, void* d_ws, size_t ws_size,
                              hipStream_t stream) {
  (void)in_sizes; (void)n_in; (void)out_size; (void)ws_size;
  const float* x  = (const float*)d_in[0];
  const float* WQ = (const float*)d_in[1];
  const float* bQ = (const float*)d_in[2];
  const float* WK = (const float*)d_in[3];
  const float* bK = (const float*)d_in[4];
  const float* WV = (const float*)d_in[5];
  const float* bV = (const float*)d_in[6];
  const float* WO = (const float*)d_in[7];
  const float* bO = (const float*)d_in[8];
  float* out = (float*)d_out;

  char* ws = (char*)d_ws;
  __bf16* xb  = (__bf16*)(ws + XB_OFF);
  __bf16* WQt = (__bf16*)(ws + WQT_OFF);
  __bf16* WKt = (__bf16*)(ws + WKT_OFF);
  __bf16* WVt = (__bf16*)(ws + WVT_OFF);
  __bf16* WOt = (__bf16*)(ws + WOT_OFF);
  __bf16* Qb  = (__bf16*)(ws + Q_OFF);
  __bf16* Kb  = (__bf16*)(ws + K_OFF);
  __bf16* Vt  = (__bf16*)(ws + VT_OFF);
  float*  Zs  = (float* )(ws + ZS_OFF);

  cast_f32_bf16_kernel<<<(B_ * S_ * D_) / 256, 256, 0, stream>>>(x, xb, B_ * S_ * D_);
  transpose_cast_kernel<<<(D_ * D_) / 256, 256, 0, stream>>>(WQ, WQt, D_, D_);
  transpose_cast_kernel<<<(D_ * D_) / 256, 256, 0, stream>>>(WK, WKt, D_, D_);
  transpose_cast_kernel<<<(D_ * D_) / 256, 256, 0, stream>>>(WV, WVt, D_, D_);
  transpose_cast_kernel<<<(DH_ * D_) / 256, 256, 0, stream>>>(WO, WOt, DH_, D_);
  zero_f32_kernel<<<(B_ * S_ * DH_) / 256, 256, 0, stream>>>(Zs, B_ * S_ * DH_);

  dim3 gq(B_ * S_ / 32, D_ / 64, 3);
  qkv_proj_kernel<<<gq, 32, 0, stream>>>(xb, WQt, WKt, WVt, bQ, bK, bV, Qb, Kb, Vt);

  dim3 ga(S_ / 32, B_ * H_);
  attn_kernel<<<ga, 32, 0, stream>>>(Qb, Kb, Vt, Zs);

  dim3 go(B_ * S_ / 16, D_ / 64);
  out_proj_kernel<<<go, 32, 0, stream>>>(Zs, WOt, bO, out);
}
